// SelfAttention_82068235092089
// MI455X (gfx1250) — matrix operand order repaired and checked by reading, not compiled, not run on hardware
//
#include <hip/hip_runtime.h>
#include <hip/hip_bf16.h>

#define UNITS 512
#define BATCH 64
#define SEQ   2048

typedef __attribute__((ext_vector_type(16))) __bf16          bf16x16;
typedef __attribute__((ext_vector_type(8)))  float           f32x8;
typedef __attribute__((ext_vector_type(8)))  unsigned short  us8;
typedef __attribute__((ext_vector_type(16))) unsigned short  us16;

__device__ __forceinline__ unsigned short f2bf(float f) {
    unsigned u = __float_as_uint(f);
    u += 0x7FFFu + ((u >> 16) & 1u);          // round-to-nearest-even
    return (unsigned short)(u >> 16);
}

// ---------------------------------------------------------------------------
// Pack Uw (512x512 f32, row-major [K][N]) into bf16 in WMMA B-fragment order:
// element gid = ((tk*32 + ntile)*32 + lane)*16 + j  holds
//   Uw[ tk*32 + (lane>>4)*16 + j ][ ntile*16 + (lane&15) ]
// so the GEMM kernel loads one 32B vector per lane, fully coalesced.
// ---------------------------------------------------------------------------
__global__ void k_pack_uw(const float* __restrict__ Uw,
                          unsigned short* __restrict__ Uwp) {
    int gid  = blockIdx.x * 256 + threadIdx.x;      // 0 .. 262143
    int j    = gid & 15;
    int lane = (gid >> 4) & 31;
    int nt   = (gid >> 9) & 31;
    int tk   = gid >> 14;
    int N = nt * 16 + (lane & 15);
    int K = tk * 32 + (lane >> 4) * 16 + j;
    Uwp[gid] = f2bf(Uw[K * UNITS + N]);
}

// ---------------------------------------------------------------------------
// w_s = relu(s_prev @ Ww + Wb)   (64x512, exact f32 — tiny)
// ---------------------------------------------------------------------------
__global__ void k_ws(const float* __restrict__ s_prev,
                     const float* __restrict__ Ww,
                     const float* __restrict__ Wb,
                     float* __restrict__ ws_g) {
    __shared__ float sp[UNITS];
    int b = blockIdx.x;
    int t = threadIdx.x;
    int n = blockIdx.y * 256 + t;
    sp[t]       = s_prev[b * UNITS + t];
    sp[t + 256] = s_prev[b * UNITS + t + 256];
    __syncthreads();
    float acc = Wb[n];
#pragma unroll 8
    for (int k = 0; k < UNITS; ++k)
        acc = fmaf(sp[k], Ww[k * UNITS + n], acc);
    ws_g[b * UNITS + n] = fmaxf(acc, 0.f);
}

// ---------------------------------------------------------------------------
// Main fused kernel: per WG (8 waves), rows = 64 seq positions of one batch.
// Wave owns 4 M-tiles x 4 N-tiles (16 f32 16x16 accumulators, 128 VGPRs).
//   u_h tile = A(64x512 bf16, LDS) x B(512x512 bf16, packed, L2-resident)
//   score[s] = sum_n tanh(w_s[b,n] + relu(u_h + Ub[n])) * Vw[n]
// u_h never touches memory; each B fragment feeds 4 WMMAs.
// ---------------------------------------------------------------------------
__global__ void __launch_bounds__(256)
k_scores(const float* __restrict__ hidden,
         const unsigned short* __restrict__ Uwp,
         const float* __restrict__ ws_g,
         const float* __restrict__ Ub,
         const float* __restrict__ Vw,
         float* __restrict__ scores) {

    __shared__ unsigned short Abuf[64 * 520];   // 64 rows x 512 bf16, stride 520 (bank spread)
    __shared__ float s_part[8][64];             // per-wave partial scores (deterministic reduce)

    const int t    = threadIdx.x;
    const int wave = t >> 5;
    const int lane = t & 31;
    const int l15  = lane & 15;
    const int half = lane >> 4;
    const int bs   = blockIdx.x;                // 0..2047
    const int b    = bs >> 5;
    const int s0   = (bs & 31) << 6;

    // ---- stage hidden tile (64 x 512 f32 -> bf16) into LDS ----------------
    const float* hbase = hidden + (size_t)(b * SEQ + s0) * UNITS;
#pragma unroll
    for (int i = 0; i < 32; ++i) {
        int linear = i * 1024 + t * 4;
        int r = linear >> 9;
        int k = linear & 511;
        float4 v = *(const float4*)(hbase + (size_t)r * UNITS + k);
        unsigned lo = (unsigned)f2bf(v.x) | ((unsigned)f2bf(v.y) << 16);
        unsigned hi = (unsigned)f2bf(v.z) | ((unsigned)f2bf(v.w) << 16);
        *(uint2*)(&Abuf[r * 520 + k]) = make_uint2(lo, hi);
    }
    __syncthreads();

    // ---- K loop: wave owns 4 M-tiles x 4 N-tiles --------------------------
    f32x8 c[4][4];
#pragma unroll
    for (int m = 0; m < 4; ++m)
#pragma unroll
        for (int nt = 0; nt < 4; ++nt)
            c[m][nt] = (f32x8){0.f, 0.f, 0.f, 0.f, 0.f, 0.f, 0.f, 0.f};

#pragma unroll
    for (int tk = 0; tk < 16; ++tk) {
        bf16x16 bfrag[4];
#pragma unroll
        for (int nt = 0; nt < 4; ++nt) {
            size_t off = ((size_t)(tk * 32 + wave * 4 + nt) * 32 + lane) * 16;
            bfrag[nt] = *(const bf16x16*)(Uwp + off);   // 32B coalesced, L2 hit
        }
#pragma unroll
        for (int m = 0; m < 4; ++m) {
            int r    = m * 16 + l15;
            int base = r * 520 + tk * 32 + half * 16;   // 16B aligned
            us8 a0 = *(const us8*)(&Abuf[base]);        // K = 16h+0..7   (+32*tk)
            us8 a1 = *(const us8*)(&Abuf[base + 8]);    // K = 16h+8..15
            us16 aw = __builtin_shufflevector(a0, a1,
                        0,1,2,3,4,5,6,7,8,9,10,11,12,13,14,15);
            union { us16 u; bf16x16 v; } cv; cv.u = aw;
            bf16x16 afrag = cv.v;
#pragma unroll
            for (int nt = 0; nt < 4; ++nt) {
                c[m][nt] = __builtin_amdgcn_wmma_f32_16x16x32_bf16(
                    false, afrag, false, bfrag[nt], (short)0, c[m][nt],
                    false, false);
            }
        }
    }

    // ---- epilogue: relu, +w_s, tanh, dot Vw  (n is per-lane constant) ----
    float part[4][8];
#pragma unroll
    for (int m = 0; m < 4; ++m)
#pragma unroll
        for (int i = 0; i < 8; ++i) part[m][i] = 0.f;

#pragma unroll
    for (int nt = 0; nt < 4; ++nt) {
        int n = wave * 64 + nt * 16 + l15;
        float wsv = ws_g[b * UNITS + n];
        float ubv = Ub[n];
        float vwv = Vw[n];
#pragma unroll
        for (int m = 0; m < 4; ++m)
#pragma unroll
            for (int i = 0; i < 8; ++i) {
                float u = fmaxf(c[m][nt][i] + ubv, 0.f);
                part[m][i] += tanhf(wsv + u) * vwv;
            }
    }

    // reduce across the 16 lanes that share a row (xor over bits 0..3)
#pragma unroll
    for (int m = 0; m < 4; ++m)
#pragma unroll
        for (int i = 0; i < 8; ++i) {
            float v = part[m][i];
#pragma unroll
            for (int d = 1; d < 16; d <<= 1) v += __shfl_xor(v, d, 32);
            part[m][i] = v;
        }
    if (l15 == 0) {
#pragma unroll
        for (int m = 0; m < 4; ++m)
#pragma unroll
            for (int i = 0; i < 8; ++i)
                s_part[wave][m * 16 + half * 8 + i] = part[m][i];   // row = 16m+8h+i
    }
    __syncthreads();

    if (t < 64) {                       // fixed-order cross-wave sum: deterministic
        float s = 0.f;
#pragma unroll
        for (int w = 0; w < 8; ++w) s += s_part[w][t];
        scores[b * SEQ + s0 + t] = s;   // (+Vb dropped: softmax-invariant)
    }
}

// ---------------------------------------------------------------------------
// Softmax over S per batch -> attention weights (written straight to d_out)
// ---------------------------------------------------------------------------
__global__ void k_softmax(const float* __restrict__ scores,
                          float* __restrict__ attn) {
    __shared__ float red[256];
    int b = blockIdx.x, t = threadIdx.x;
    float v[8];
    float mx = -3.4e38f;
#pragma unroll
    for (int j = 0; j < 8; ++j) {
        v[j] = scores[b * SEQ + j * 256 + t];
        mx = fmaxf(mx, v[j]);
    }
    red[t] = mx; __syncthreads();
    for (int s = 128; s > 0; s >>= 1) {
        if (t < s) red[t] = fmaxf(red[t], red[t + s]);
        __syncthreads();
    }
    mx = red[0]; __syncthreads();
    float sum = 0.f;
#pragma unroll
    for (int j = 0; j < 8; ++j) { v[j] = expf(v[j] - mx); sum += v[j]; }
    red[t] = sum; __syncthreads();
    for (int s = 128; s > 0; s >>= 1) {
        if (t < s) red[t] += red[t + s];
        __syncthreads();
    }
    float inv = 1.0f / red[0];
#pragma unroll
    for (int j = 0; j < 8; ++j)
        attn[b * SEQ + j * 256 + t] = v[j] * inv;
}

// ---------------------------------------------------------------------------
// context = sum_s attn[b,s] * hidden[b,s,:]  — two-phase deterministic reduce
// ---------------------------------------------------------------------------
__global__ void k_ctx_part(const float* __restrict__ hidden,
                           const float* __restrict__ attn,
                           float* __restrict__ partial) {
    int b = blockIdx.x, ch = blockIdx.y, t = threadIdx.x;
    const float* hb = hidden + (size_t)(b * SEQ + ch * 256) * UNITS;
    const float* wb = attn + b * SEQ + ch * 256;
    float a0 = 0.f, a1 = 0.f;
    for (int s = 0; s < 256; ++s) {
        float w = wb[s];
        a0 = fmaf(w, hb[(size_t)s * UNITS + t], a0);
        a1 = fmaf(w, hb[(size_t)s * UNITS + t + 256], a1);
    }
    float* p = partial + (size_t)(b * 8 + ch) * UNITS;
    p[t] = a0;
    p[t + 256] = a1;
}

__global__ void k_ctx_red(const float* __restrict__ partial,
                          float* __restrict__ ctx) {
    int gid = blockIdx.x * 256 + threadIdx.x;   // 0..32767
    int b = gid >> 9, u = gid & 511;
    float s = 0.f;
#pragma unroll
    for (int ch = 0; ch < 8; ++ch)
        s += partial[(size_t)(b * 8 + ch) * UNITS + u];
    ctx[gid] = s;
}

// ---------------------------------------------------------------------------
extern "C" void kernel_launch(void* const* d_in, const int* in_sizes, int n_in,
                              void* d_out, int out_size, void* d_ws, size_t ws_size,
                              hipStream_t stream) {
    (void)in_sizes; (void)n_in; (void)out_size; (void)ws_size;
    const float* s_prev = (const float*)d_in[0];
    const float* hidden = (const float*)d_in[1];
    const float* Ww     = (const float*)d_in[2];
    const float* Wb     = (const float*)d_in[3];
    const float* Uw     = (const float*)d_in[4];
    const float* Ub     = (const float*)d_in[5];
    const float* Vw     = (const float*)d_in[6];
    // d_in[7] = Vb: softmax-invariant, not needed for either output.

    float* out  = (float*)d_out;
    float* ctx  = out;                         // (64,512)   = 32768 floats
    float* attn = out + BATCH * UNITS;         // (64,2048)  = 131072 floats

    float* ws      = (float*)d_ws;
    float* scores  = ws;                                   // 131072 f32
    float* ws_g    = ws + 131072;                          // 32768 f32
    unsigned short* Uwp = (unsigned short*)(ws + 163840);  // 262144 bf16 (512KB)
    float* partial = ws + 163840 + 131072;                 // 262144 f32

    k_pack_uw<<<1024, 256, 0, stream>>>(Uw, Uwp);
    k_ws<<<dim3(64, 2), 256, 0, stream>>>(s_prev, Ww, Wb, ws_g);
    k_scores<<<BATCH * (SEQ / 64), 256, 0, stream>>>(hidden, Uwp, ws_g, Ub, Vw, scores);
    k_softmax<<<BATCH, 256, 0, stream>>>(scores, attn);
    k_ctx_part<<<dim3(BATCH, 8), 256, 0, stream>>>(hidden, attn, partial);
    k_ctx_red<<<128, 256, 0, stream>>>(partial, ctx);
}
